// PointerNetwork_89721866813625
// MI455X (gfx1250) — matrix operand order repaired and checked by reading, not compiled, not run on hardware
//
#include <hip/hip_runtime.h>
#include <hip/hip_bf16.h>
#include <math.h>
#include <stdint.h>

#define HDIM   512
#define BN     64
#define WLEN   512
#define KSTEPS 8
#define TWOH   1024
#define THREEH 1536
#define RROWS  ((WLEN + 1) * BN)        // 513*64 = 32832 rows (s-major, n inner)
#define NEG_FACTOR (1.0f - 1.0e10f)
#define KCHUNK 16                        // k-tiles staged per LDS chunk (32 KB)

typedef __bf16 bf16_t;
typedef bf16_t v16bf __attribute__((ext_vector_type(16)));
typedef float  v8f   __attribute__((ext_vector_type(8)));
typedef unsigned int u32;

#if defined(__gfx1250__) && __has_builtin(__builtin_amdgcn_tensor_load_to_lds)
#define USE_TDM 1
#else
#define USE_TDM 0
#endif

// ---------------------------------------------------------------------------
// CDNA5 WMMA fragment coordinate maps (ISA 7.12.2, wave32)
// A (16x32 bf16): lanes 0-15 -> K-half 0, lanes 16-31 -> K-half 1 (M = L&15).
//   VGPR0-3: K = half*8 + {0..7}; VGPR4-7: K = 16 + half*8 + {0..7}.
// B (32x16 bf16): col = L&15, K = 8*(L>>4) + e for e<8, 16 + 8*(L>>4) + (e-8) for e>=8 (same as A).
// C/D (16x16 f32): col = L&15, row = 8*(L>>4) + vgpr j.
// ---------------------------------------------------------------------------
__device__ __forceinline__ void a_coords(int lane, int e, int& m, int& k) {
  m = lane & 15;
  int half = lane >> 4;
  int vg = e >> 1;
  k = ((vg >> 2) << 4) | (half << 3) | ((vg & 3) << 1) | (e & 1);
}

// Swizzle masked context_new (r = s*64+n, row s=512 zeros) into A fragments.
__global__ void k_swizzleA_ctx(const float* __restrict__ ctx,
                               const int* __restrict__ lens,
                               bf16_t* __restrict__ dst) {
  const int numKt = TWOH / 32;
  long long idx = (long long)blockIdx.x * blockDim.x + threadIdx.x;
  const long long total = (long long)(RROWS / 16) * numKt * 512;
  if (idx >= total) return;
  int e = (int)(idx & 15);
  int lane = (int)((idx >> 4) & 31);
  long long t = idx >> 9;
  int kt = (int)(t % numKt);
  int tm = (int)(t / numKt);
  int m, k; a_coords(lane, e, m, k);
  int r = tm * 16 + m;
  int s = r >> 6;
  int n = r & 63;
  float val = 0.0f;
  if (s < WLEN) {
    float c = ctx[((long long)n * WLEN + s) * TWOH + kt * 32 + k];
    float f = (s >= lens[n] + 1) ? NEG_FACTOR : 1.0f;
    val = c * f;
  }
  dst[idx] = (bf16_t)val;
}

// Generic A swizzle from f32 row-major [Mrows x Kdim].
__global__ void k_swizzleA(const float* __restrict__ src, bf16_t* __restrict__ dst,
                           int Mrows, int Kdim) {
  int numKt = Kdim / 32;
  long long idx = (long long)blockIdx.x * blockDim.x + threadIdx.x;
  long long total = (long long)(Mrows / 16) * numKt * 512;
  if (idx >= total) return;
  int e = (int)(idx & 15);
  int lane = (int)((idx >> 4) & 31);
  long long t = idx >> 9;
  int kt = (int)(t % numKt);
  int tm = (int)(t / numKt);
  int m, k; a_coords(lane, e, m, k);
  dst[idx] = (bf16_t)src[(long long)(tm * 16 + m) * Kdim + kt * 32 + k];
}

// B swizzle: weight row-major [Nout x Kdim]; B[k][col] = Wt[col][k].
__global__ void k_swizzleB(const float* __restrict__ wt, bf16_t* __restrict__ dst,
                           int Nout, int Kdim) {
  int numKt = Kdim / 32;
  long long idx = (long long)blockIdx.x * blockDim.x + threadIdx.x;
  long long total = (long long)(Nout / 16) * numKt * 512;
  if (idx >= total) return;
  int e = (int)(idx & 15);
  int lane = (int)((idx >> 4) & 31);
  long long t = idx >> 9;
  int kt = (int)(t % numKt);
  int tn = (int)(t / numKt);
  int col = tn * 16 + (lane & 15);
  int k = kt * 32 + ((e >> 3) << 4) + (lane >> 4) * 8 + (e & 7);
  dst[idx] = (bf16_t)wt[(long long)col * Kdim + k];
}

#if USE_TDM
// TDM: DMA a 2-row tile (2 column-tiles x KCHUNK k-tiles, 16 KB each) into LDS.
// D# per ISA 8.3/8.4: count=1, type=2(image), data_size=3 (8 bytes),
// tile 2048x2, tensor_dim0_stride jumps between the two column tiles.
// Toolchain here uses the 6-arg builtin: (g0, g1, g2, g3, g4, cpol).
__device__ __forceinline__ void tdm_load_b_panel(const bf16_t* gsrc, u32 lds_off,
                                                 u32 tile_stride_e8) {
  typedef u32 u32v4 __attribute__((ext_vector_type(4)));
  typedef int i32v8 __attribute__((ext_vector_type(8)));
  typedef int i32v4 __attribute__((ext_vector_type(4)));
  unsigned long long ga = (unsigned long long)(uintptr_t)gsrc;
  const u32 dim0 = (KCHUNK * 1024u) / 8u;            // 2048 8-byte elems / row
  u32v4 g0;
  g0[0] = 1u;                                        // count = 1 descriptor
  g0[1] = lds_off;                                   // lds_addr (bytes)
  g0[2] = (u32)(ga & 0xffffffffull);                 // global_addr[31:0]
  g0[3] = (u32)((ga >> 32) & 0x1ffffffull) | (2u << 30);  // addr[56:32] | type=2
  i32v8 g1;
  g1[0] = (int)(3u << 16);                           // data_size = 3 (8B), mask=0
  g1[1] = (int)((dim0 & 0xffffu) << 16);             // tensor_dim0[15:0]
  g1[2] = (int)((dim0 >> 16) | (2u << 16));          // dim0[31:16] | tensor_dim1=2
  g1[3] = (int)((dim0 & 0xffffu) << 16);             // tile_dim0 = 2048
  g1[4] = 2;                                         // tile_dim1 = 2, tile_dim2=0
  g1[5] = (int)tile_stride_e8;                       // tensor_dim0_stride[31:0]
  g1[6] = 0;                                         // stride hi | dim1_stride lo
  g1[7] = 0;
  i32v4 z4 = {0, 0, 0, 0};
  i32v8 z8 = {0, 0, 0, 0, 0, 0, 0, 0};
  __builtin_amdgcn_tensor_load_to_lds(g0, g1, z4, z4, z8, 0);
}
#endif

// C[M x ldc] f32 = A_swz x B_swz. 2x2 register-blocked: one wave owns a 32x32
// output; all 8 waves of a block share one 32-column B panel staged in LDS
// (TDM async-tensor DMA when available).
__global__ void __launch_bounds__(256) k_gemm2x2(
    const bf16_t* __restrict__ A, const bf16_t* __restrict__ B,
    float* __restrict__ C, int numTmB, int numTnB, int numKt, int ldc) {
  __shared__ bf16_t ldsB[2 * KCHUNK * 512];          // 32 KB B panel chunk
  int lane = threadIdx.x & 31;
  int wid = threadIdx.x >> 5;
  int cb = blockIdx.x % numTnB;
  int rb = (blockIdx.x / numTnB) * 8 + wid;
  bool active = rb < numTmB;                         // wave-uniform

  int tm0 = 2 * rb;
  int tn0 = 2 * cb;
  const bf16_t* ap0 = A + (long long)tm0 * numKt * 512 + lane * 16;
  const bf16_t* ap1 = ap0 + (long long)numKt * 512;
  v8f acc00 = {0.f, 0.f, 0.f, 0.f, 0.f, 0.f, 0.f, 0.f};
  v8f acc01 = acc00, acc10 = acc00, acc11 = acc00;

  for (int kc = 0; kc < numKt; kc += KCHUNK) {
    // ---- stage B panel chunk (2 tiles x 16 KB) into LDS ----
#if USE_TDM
    if (wid == 0) {
      tdm_load_b_panel(B + ((long long)tn0 * numKt + kc) * 512,
                       (u32)(uintptr_t)(void*)&ldsB[0],
                       (u32)(numKt * 128));          // tile gap in 8B units
      __builtin_amdgcn_s_wait_tensorcnt(0);
    }
#else
    {
      const uint4* s0 = (const uint4*)(B + ((long long)tn0 * numKt + kc) * 512);
      const uint4* s1 = (const uint4*)(B + ((long long)(tn0 + 1) * numKt + kc) * 512);
      uint4* l0 = (uint4*)&ldsB[0];
      uint4* l1 = (uint4*)&ldsB[KCHUNK * 512];
      for (int i = threadIdx.x; i < KCHUNK * 64; i += 256) {  // 1024 x uint4
        l0[i] = s0[i];
        l1[i] = s1[i];
      }
    }
#endif
    __syncthreads();

    if (active) {
      const bf16_t* lb = ldsB + lane * 16;
#pragma unroll 4
      for (int ktl = 0; ktl < KCHUNK; ++ktl) {
        v16bf a0 = *(const v16bf*)ap0; ap0 += 512;
        v16bf a1 = *(const v16bf*)ap1; ap1 += 512;
        __builtin_prefetch(ap0, 0, 1);               // global_prefetch_b8
        __builtin_prefetch(ap1, 0, 1);
        v16bf b0 = *(const v16bf*)(lb + ktl * 512);
        v16bf b1 = *(const v16bf*)(lb + (KCHUNK + ktl) * 512);
        acc00 = __builtin_amdgcn_wmma_f32_16x16x32_bf16(false, a0, false, b0,
                                                        (short)0, acc00, false, false);
        acc01 = __builtin_amdgcn_wmma_f32_16x16x32_bf16(false, a0, false, b1,
                                                        (short)0, acc01, false, false);
        acc10 = __builtin_amdgcn_wmma_f32_16x16x32_bf16(false, a1, false, b0,
                                                        (short)0, acc10, false, false);
        acc11 = __builtin_amdgcn_wmma_f32_16x16x32_bf16(false, a1, false, b1,
                                                        (short)0, acc11, false, false);
      }
    }
    __syncthreads();
  }

  if (!active) return;
  int ncol = lane & 15;
  int mh = (lane >> 4) * 8;
#pragma unroll
  for (int j = 0; j < 8; ++j) {
    long long r0 = (long long)(tm0 * 16 + mh + j) * ldc;
    long long r1 = (long long)((tm0 + 1) * 16 + mh + j) * ldc;
    C[r0 + tn0 * 16 + ncol]        = acc00[j];
    C[r0 + (tn0 + 1) * 16 + ncol]  = acc01[j];
    C[r1 + tn0 * 16 + ncol]        = acc10[j];
    C[r1 + (tn0 + 1) * 16 + ncol]  = acc11[j];
  }
}

// scores[r] = sum_h tanh(ctx_t[r][h] + hW[n][h]) * v[h]; one wave per row.
__global__ void k_score(const float* __restrict__ ctx_t, const float* __restrict__ hW,
                        const float* __restrict__ vvec, float* __restrict__ scores) {
  int lane = threadIdx.x & 31;
  int wave = blockIdx.x * (blockDim.x >> 5) + (threadIdx.x >> 5);
  if (wave >= RROWS) return;
  int n = wave & 63;
  const float* cr = ctx_t + (long long)wave * HDIM;
  const float* hr = hW + (long long)n * HDIM;
  float sum = 0.f;
  for (int h = lane; h < HDIM; h += 32)
    sum += tanhf(cr[h] + hr[h]) * vvec[h];
  for (int off = 16; off > 0; off >>= 1)
    sum += __shfl_xor(sum, off, 32);
  if (lane == 0) scores[wave] = sum;
}

// softmax over s for each column n; writes beta slice of d_out and beta_ws.
__global__ void k_softmax(const float* __restrict__ scores, float* __restrict__ beta_ws,
                          float* __restrict__ out, int step) {
  __shared__ float red[256];
  int n = blockIdx.x;
  int tid = threadIdx.x;
  float m = -3.4e38f;
  for (int s = tid; s <= WLEN; s += 256) m = fmaxf(m, scores[s * BN + n]);
  red[tid] = m; __syncthreads();
  for (int o = 128; o > 0; o >>= 1) {
    if (tid < o) red[tid] = fmaxf(red[tid], red[tid + o]);
    __syncthreads();
  }
  float mx = red[0]; __syncthreads();
  float sum = 0.f;
  for (int s = tid; s <= WLEN; s += 256) sum += expf(scores[s * BN + n] - mx);
  red[tid] = sum; __syncthreads();
  for (int o = 128; o > 0; o >>= 1) {
    if (tid < o) red[tid] += red[tid + o];
    __syncthreads();
  }
  float inv = 1.f / red[0];
  for (int s = tid; s <= WLEN; s += 256) {
    float b = expf(scores[s * BN + n] - mx) * inv;
    beta_ws[s * BN + n] = b;
    out[((long long)n * KSTEPS + step) * (WLEN + 1) + s] = b;
  }
}

// aligned[n][d] = sum_s beta[s,n] * (context[n][s][d] * factor(s,n))
__global__ void k_aligned(const float* __restrict__ ctx, const int* __restrict__ lens,
                          const float* __restrict__ beta, float* __restrict__ aligned) {
  int n = blockIdx.y;
  int d = blockIdx.x * 256 + threadIdx.x;
  int len = lens[n];
  const float* cp = ctx + (long long)n * WLEN * TWOH + d;
  float acc = 0.f;
  for (int s = 0; s < WLEN; ++s) {
    float f = (s >= len + 1) ? NEG_FACTOR : 1.0f;
    acc += beta[s * BN + n] * (cp[(long long)s * TWOH] * f);
  }
  aligned[(long long)n * TWOH + d] = acc;
}

// torch GRUCell update, in place on h.
__global__ void k_gru(const float* __restrict__ gi, const float* __restrict__ gh,
                      const float* __restrict__ b_ih, const float* __restrict__ b_hh,
                      float* __restrict__ h) {
  int i = blockIdx.x * blockDim.x + threadIdx.x;
  if (i >= BN * HDIM) return;
  int n = i >> 9;
  int j = i & 511;
  const float* gip = gi + (long long)n * THREEH;
  const float* ghp = gh + (long long)n * THREEH;
  float ir = gip[j] + b_ih[j];
  float iz = gip[HDIM + j] + b_ih[HDIM + j];
  float ic = gip[2 * HDIM + j] + b_ih[2 * HDIM + j];
  float hr = ghp[j] + b_hh[j];
  float hz = ghp[HDIM + j] + b_hh[HDIM + j];
  float hc = ghp[2 * HDIM + j] + b_hh[2 * HDIM + j];
  float r = 1.f / (1.f + expf(-(ir + hr)));
  float z = 1.f / (1.f + expf(-(iz + hz)));
  float nc = tanhf(ic + r * hc);
  h[i] = (1.f - z) * nc + z * h[i];
}

__global__ void k_zero(float* __restrict__ p, int count) {
  int i = blockIdx.x * blockDim.x + threadIdx.x;
  if (i < count) p[i] = 0.f;
}

static inline void launch_gemm(const bf16_t* A, const bf16_t* B, float* C,
                               int numTm, int numTn, int numKt, int ldc,
                               hipStream_t stream) {
  int numTmB = numTm / 2, numTnB = numTn / 2;
  int rowBlocks = (numTmB + 7) / 8;
  k_gemm2x2<<<rowBlocks * numTnB, 256, 0, stream>>>(A, B, C, numTmB, numTnB,
                                                    numKt, ldc);
}

extern "C" void kernel_launch(void* const* d_in, const int* in_sizes, int n_in,
                              void* d_out, int out_size, void* d_ws, size_t ws_size,
                              hipStream_t stream) {
  (void)in_sizes; (void)n_in; (void)out_size; (void)ws_size;
  const float* ctx  = (const float*)d_in[0];
  const int*   lens = (const int*)d_in[1];
  // d_in[2] pointer_answers: only its K dimension matters (fixed at 8)
  const float* V_w  = (const float*)d_in[3];
  const float* W_w  = (const float*)d_in[4];
  const float* vvec = (const float*)d_in[5];
  const float* w_ih = (const float*)d_in[6];
  const float* w_hh = (const float*)d_in[7];
  const float* b_ih = (const float*)d_in[8];
  const float* b_hh = (const float*)d_in[9];
  float* out = (float*)d_out;

  char* ws = (char*)d_ws;
  size_t off = 0;
  auto alloc = [&](size_t bytes) -> char* {
    char* p = ws + off;
    off += (bytes + 255) & ~(size_t)255;
    return p;
  };
  bf16_t* a_ctx    = (bf16_t*)alloc((size_t)RROWS * TWOH * 2);   // ~67 MB
  bf16_t* b_vw     = (bf16_t*)alloc((size_t)TWOH * HDIM * 2);
  bf16_t* b_wih    = (bf16_t*)alloc((size_t)TWOH * THREEH * 2);
  bf16_t* b_whh    = (bf16_t*)alloc((size_t)HDIM * THREEH * 2);
  bf16_t* b_ww     = (bf16_t*)alloc((size_t)HDIM * HDIM * 2);
  float*  ctx_t    = (float*)alloc((size_t)RROWS * HDIM * 4);    // ~67 MB
  float*  hW       = (float*)alloc((size_t)BN * HDIM * 4);
  float*  hbuf     = (float*)alloc((size_t)BN * HDIM * 4);
  bf16_t* h_swz    = (bf16_t*)alloc((size_t)BN * HDIM * 2);
  float*  alignedv = (float*)alloc((size_t)BN * TWOH * 4);
  bf16_t* al_swz   = (bf16_t*)alloc((size_t)BN * TWOH * 2);
  float*  gi       = (float*)alloc((size_t)BN * THREEH * 4);
  float*  gh       = (float*)alloc((size_t)BN * THREEH * 4);
  float*  scores   = (float*)alloc((size_t)RROWS * 4);
  float*  beta_ws  = (float*)alloc((size_t)RROWS * 4);

  // h0 = 0
  k_zero<<<(BN * HDIM + 255) / 256, 256, 0, stream>>>(hbuf, BN * HDIM);

  // One-time swizzles into WMMA fragment layouts
  {
    long long tot = (long long)(RROWS / 16) * (TWOH / 32) * 512;
    k_swizzleA_ctx<<<(int)((tot + 255) / 256), 256, 0, stream>>>(ctx, lens, a_ctx);
  }
  {
    long long tot = (long long)(HDIM / 16) * (TWOH / 32) * 512;
    k_swizzleB<<<(int)((tot + 255) / 256), 256, 0, stream>>>(V_w, b_vw, HDIM, TWOH);
  }
  {
    long long tot = (long long)(THREEH / 16) * (TWOH / 32) * 512;
    k_swizzleB<<<(int)((tot + 255) / 256), 256, 0, stream>>>(w_ih, b_wih, THREEH, TWOH);
  }
  {
    long long tot = (long long)(THREEH / 16) * (HDIM / 32) * 512;
    k_swizzleB<<<(int)((tot + 255) / 256), 256, 0, stream>>>(w_hh, b_whh, THREEH, HDIM);
  }
  {
    long long tot = (long long)(HDIM / 16) * (HDIM / 32) * 512;
    k_swizzleB<<<(int)((tot + 255) / 256), 256, 0, stream>>>(W_w, b_ww, HDIM, HDIM);
  }

  // ctx_t = context_new @ V_w^T  (34.4 GFLOP, bf16 WMMA -> f32)
  launch_gemm(a_ctx, b_vw, ctx_t, RROWS / 16, HDIM / 16, TWOH / 32, HDIM, stream);

  for (int step = 0; step < KSTEPS; ++step) {
    // h -> bf16 fragments
    {
      long long tot = (long long)(BN / 16) * (HDIM / 32) * 512;
      k_swizzleA<<<(int)((tot + 255) / 256), 256, 0, stream>>>(hbuf, h_swz, BN, HDIM);
    }
    // hW = h @ W_w^T
    launch_gemm(h_swz, b_ww, hW, BN / 16, HDIM / 16, HDIM / 32, HDIM, stream);
    // scores = tanh(ctx_t + hW) . v
    k_score<<<(RROWS * 32 + 255) / 256, 256, 0, stream>>>(ctx_t, hW, vvec, scores);
    // softmax over positions; emit this step's beta slice of the output
    k_softmax<<<BN, 256, 0, stream>>>(scores, beta_ws, out, step);
    // aligned = beta^T-weighted context_new
    {
      dim3 g(TWOH / 256, BN);
      k_aligned<<<g, 256, 0, stream>>>(ctx, lens, beta_ws, alignedv);
    }
    // aligned -> bf16 fragments
    {
      long long tot = (long long)(BN / 16) * (TWOH / 32) * 512;
      k_swizzleA<<<(int)((tot + 255) / 256), 256, 0, stream>>>(alignedv, al_swz, BN, TWOH);
    }
    // gi = aligned @ w_ih^T ; gh = h @ w_hh^T
    launch_gemm(al_swz, b_wih, gi, BN / 16, THREEH / 16, TWOH / 32, THREEH, stream);
    launch_gemm(h_swz, b_whh, gh, BN / 16, THREEH / 16, HDIM / 32, THREEH, stream);
    // GRU cell update (in place)
    k_gru<<<(BN * HDIM + 255) / 256, 256, 0, stream>>>(gi, gh, b_ih, b_hh, hbuf);
  }
}
